// EnNet_15040975470806
// MI455X (gfx1250) — matrix operand order repaired and checked by reading, not compiled, not run on hardware
//
#include <hip/hip_runtime.h>
#include <hip/hip_bf16.h>
#include <math.h>

// ---------------- dims ----------------
#define BATCH   2
#define NN      1024          // nodes per batch
#define NB      (BATCH*NN)    // 2048 total nodes
#define DMODEL  64
#define HEADS   8
#define DHEAD   128
#define INNER   (HEADS*DHEAD) // 1024
#define TOPK    30
#define EHID    18
#define CHID    32
#define FFHID   256
#define NLAYER  4
#define QKSCALE 0.08838834764831845f   // 128^-0.5

typedef __bf16 bf16_t;
typedef __attribute__((ext_vector_type(16))) __bf16 v16bf;
typedef __attribute__((ext_vector_type(8)))  float  v8f;
typedef __attribute__((ext_vector_type(4)))  unsigned int u32x4;
typedef __attribute__((ext_vector_type(8)))  int i32x8;
typedef __attribute__((ext_vector_type(4)))  int i32x4;

#if defined(__has_builtin)
#  if __has_builtin(__builtin_amdgcn_tensor_load_to_lds)
#    define HAVE_TDM 1
#  else
#    define HAVE_TDM 0
#  endif
#else
#  define HAVE_TDM 0
#endif

__device__ __forceinline__ float gelu_exact(float x) {
    return 0.5f * x * (1.0f + erff(x * 0.70710678118654752440f));
}

#if HAVE_TDM
// 2D tile load global->LDS via Tensor Data Mover.
// tile_w elements contiguous per row, tile_h rows, row stride = stride_elems
// (all in 2-byte bf16 elements). Tensor dims set huge so OOB never triggers.
__device__ __forceinline__ void tdm_load_2d(unsigned lds_off, const void* gptr,
                                            unsigned tile_w, unsigned tile_h,
                                            unsigned long long stride_elems) {
    unsigned long long ga = (unsigned long long)gptr;
    u32x4 g0;
    g0[0] = 1u;                                        // count=1, user mode, no gather
    g0[1] = lds_off;                                   // LDS byte address
    g0[2] = (unsigned)(ga & 0xffffffffu);              // global addr [31:0]
    g0[3] = (unsigned)((ga >> 32) & 0x01ffffffu)       // global addr [56:32]
          | (2u << 30);                                // type=2 ("image")
    const unsigned long long td0 = 0x7fffffffull;      // tensor_dim0 (OOB off)
    const unsigned long long td1 = 0x7fffffffull;      // tensor_dim1 (OOB off)
    i32x8 g1;
    g1[0] = (int)(1u << 16);                           // wg_mask=0, data_size=1 (2B)
    g1[1] = (int)((td0 & 0xffffu) << 16);              // [63:48] tensor_dim0 lo16
    g1[2] = (int)(((td0 >> 16) & 0xffffu) | ((td1 & 0xffffu) << 16));
    g1[3] = (int)(((td1 >> 16) & 0xffffu) | ((tile_w & 0xffffu) << 16)); // tile_dim0
    g1[4] = (int)(tile_h & 0xffffu);                   // tile_dim1, tile_dim2=0
    g1[5] = (int)(stride_elems & 0xffffffffull);       // dim0 stride lo32
    g1[6] = (int)((stride_elems >> 32) & 0xffffull);   // dim0 stride hi16
    g1[7] = 0;
    i32x4 z4 = {0,0,0,0};
#if __clang_major__ >= 23
    i32x8 z8 = {0,0,0,0,0,0,0,0};
    __builtin_amdgcn_tensor_load_to_lds(g0, g1, z4, z4, z8, 0);
#else
    __builtin_amdgcn_tensor_load_to_lds(g0, g1, z4, z4, 0);
#endif
}
#endif

// ---------------- feature encoder: x = relu([sin(f),cos(f)] @ W + b) ----------------
__global__ void featenc_kernel(const float* __restrict__ feats,
                               const float* __restrict__ W,   // [6,64]
                               const float* __restrict__ b,   // [64]
                               float* __restrict__ x) {       // [NB,64]
    int n = blockIdx.x;
    int d = threadIdx.x;
    float f0 = feats[n*3+0], f1 = feats[n*3+1], f2 = feats[n*3+2];
    float s = b[d]
            + sinf(f0)*W[0*DMODEL+d] + sinf(f1)*W[1*DMODEL+d] + sinf(f2)*W[2*DMODEL+d]
            + cosf(f0)*W[3*DMODEL+d] + cosf(f1)*W[4*DMODEL+d] + cosf(f2)*W[5*DMODEL+d];
    x[n*DMODEL+d] = fmaxf(s, 0.0f);
}

// ---------------- simple copies / converts ----------------
__global__ void copy_f32_kernel(const float* __restrict__ s, float* __restrict__ d, int n) {
    int i = blockIdx.x*256 + threadIdx.x;
    if (i < n) d[i] = s[i];
}
// fp32 [Kt,Nc] row-major -> bf16 TRANSPOSED [Nc,Kt] (K contiguous per output column)
__global__ void f2bf_t_kernel(const float* __restrict__ s, bf16_t* __restrict__ d,
                              int Kt, int Nc) {
    int i = blockIdx.x*256 + threadIdx.x;
    if (i < Kt*Nc) {
        int k = i / Nc, n = i - k*Nc;
        d[(size_t)n*Kt + k] = (bf16_t)s[i];
    }
}

// ---------------- KNN: one wave per (b,i) row, LDS distance table ----------------
__global__ void knn_kernel(const float* __restrict__ coors,  // [NB,3]
                           int* __restrict__ idxout) {       // [NB,TOPK]
    __shared__ float dist[NN];
    int row  = blockIdx.x;            // b*NN + i
    int i    = row & (NN-1);
    int base = row & ~(NN-1);         // b*NN
    int lane = threadIdx.x;
    float cx = coors[row*3+0], cy = coors[row*3+1], cz = coors[row*3+2];
    for (int t = 0; t < NN/32; ++t) {
        int j = t*32 + lane;
        const float* cj = coors + (size_t)(base + j)*3;
        float dx = cx - cj[0], dy = cy - cj[1], dz = cz - cj[2];
        float d  = sqrtf(fmaxf(dx*dx + dy*dy + dz*dz, 1e-12f));
        if (j == i) d = -1.0f;        // self always kept (matches ranking diag = -1)
        dist[j] = d;
    }
    __syncthreads();
    for (int s = 0; s < TOPK; ++s) {
        float best = 1e30f; int bidx = 0x7fffffff;
        for (int t = 0; t < NN/32; ++t) {
            int j = t*32 + lane;
            float d = dist[j];
            if (d < best || (d == best && j < bidx)) { best = d; bidx = j; }
        }
        for (int off = 16; off > 0; off >>= 1) {
            float ob = __shfl_down(best, off, 32);
            int   oi = __shfl_down(bidx, off, 32);
            if (ob < best || (ob == best && oi < bidx)) { best = ob; bidx = oi; }
        }
        bidx = __shfl(bidx, 0, 32);
        if (lane == 0) { idxout[row*TOPK + s] = bidx; dist[bidx] = 2e30f; }
        __syncthreads();
    }
}

// ---------------- LayerNorm (D=64) -> bf16, one wave per node ----------------
__global__ void ln_kernel(const float* __restrict__ x,
                          const float* __restrict__ g, const float* __restrict__ b,
                          bf16_t* __restrict__ out) {
    int n = blockIdx.x, lane = threadIdx.x;
    float v0 = x[n*DMODEL + lane], v1 = x[n*DMODEL + 32 + lane];
    float s = v0 + v1;
    for (int o = 16; o > 0; o >>= 1) s += __shfl_xor(s, o, 32);
    float mean = s * (1.0f/64.0f);
    float d0 = v0 - mean, d1 = v1 - mean;
    float vs = d0*d0 + d1*d1;
    for (int o = 16; o > 0; o >>= 1) vs += __shfl_xor(vs, o, 32);
    float inv = rsqrtf(vs * (1.0f/64.0f) + 1e-5f);
    out[n*DMODEL + lane]      = (bf16_t)(d0*inv*g[lane]      + b[lane]);
    out[n*DMODEL + 32 + lane] = (bf16_t)(d1*inv*g[lane+32]   + b[lane+32]);
}

// ---------------- WMMA GEMM with TDM-staged, double-buffered LDS tiles ----------------
// C[M=2048,Nc] = A[M,KT](bf16, row-major) @ W (bf16, stored TRANSPOSED [Nc,KT]).
// One wave per 16x16 tile. Pipeline: TDM chunk c+1 -> LDS while consuming chunk c
// (s_wait_tensorcnt 2). Fragments read as 2x b128 LDS loads each.
enum { GF_BIAS = 1, GF_RESID = 2, GF_ACT = 4, GF_OUTF = 8, GF_OUTB = 16 };

union frag16 { v16bf v; uint4 q[2]; };

template<int KT, int FLAGS>
__global__ void wmma_gemm_kernel(const bf16_t* __restrict__ A,
                                 const bf16_t* __restrict__ Wt,   // [Nc,KT]
                                 const float* __restrict__ bias,
                                 const float* __restrict__ resid,
                                 float* __restrict__ Cf,
                                 bf16_t* __restrict__ Cb,
                                 int Nc) {
    constexpr int KC = 64;              // K-chunk (2 WMMA steps)
    constexpr int CHUNKS = KT / KC;
    int lane  = threadIdx.x;
    int ntile = blockIdx.x, mtile = blockIdx.y;
    int mn    = lane & 15;              // row (A-frag) / col (B,C-frag)
    int half  = lane >> 4;
    v8f acc = {};

#if HAVE_TDM
    __shared__ alignas(16) bf16_t ldsA[2][16*KC];   // [m][k]
    __shared__ alignas(16) bf16_t ldsB[2][16*KC];   // [n][k]  (K contiguous)
    const bf16_t* Abase = A  + (size_t)(mtile*16)*KT;
    const bf16_t* Bbase = Wt + (size_t)(ntile*16)*KT;
    // prime chunk 0 (both tiles are 16 rows x KC contiguous elements, stride KT)
    tdm_load_2d((unsigned)(unsigned long long)(uintptr_t)&ldsA[0][0],
                Abase, KC, 16, (unsigned long long)KT);
    tdm_load_2d((unsigned)(unsigned long long)(uintptr_t)&ldsB[0][0],
                Bbase, KC, 16, (unsigned long long)KT);
    for (int c = 0; c < CHUNKS; ++c) {
        int buf = c & 1;
        if (c + 1 < CHUNKS) {
            tdm_load_2d((unsigned)(unsigned long long)(uintptr_t)&ldsA[buf^1][0],
                        Abase + (c+1)*KC, KC, 16, (unsigned long long)KT);
            tdm_load_2d((unsigned)(unsigned long long)(uintptr_t)&ldsB[buf^1][0],
                        Bbase + (c+1)*KC, KC, 16, (unsigned long long)KT);
            __builtin_amdgcn_s_wait_tensorcnt(2);   // chunk c's two TDMs are done
        } else {
            __builtin_amdgcn_s_wait_tensorcnt(0);
        }
        asm volatile("" ::: "memory");
        #pragma unroll
        for (int kk = 0; kk < 2; ++kk) {
            int k0 = kk*32;
            // A fragment: two contiguous 8-elem runs at k0+half*8 and k0+16+half*8
            const bf16_t* ar = &ldsA[buf][mn*KC + k0];
            frag16 af;
            af.q[0] = *(const uint4*)(ar + half*8);
            af.q[1] = *(const uint4*)(ar + 16 + half*8);
            // B fragment: two contiguous 8-elem runs at column mn, k0+half*8 and k0+16+half*8
            const bf16_t* br = &ldsB[buf][mn*KC + k0];
            frag16 bfr;
            bfr.q[0] = *(const uint4*)(br + half*8);
            bfr.q[1] = *(const uint4*)(br + 16 + half*8);
            acc = __builtin_amdgcn_wmma_f32_16x16x32_bf16(false, af.v, false, bfr.v,
                                                          (short)0, acc, false, false);
        }
    }
#else
    // fallback: direct global-load fragments (both operands K-contiguous)
    for (int k0 = 0; k0 < KT; k0 += 32) {
        const bf16_t* ar = A + (size_t)(mtile*16 + mn)*KT + k0;
        frag16 af;
        af.q[0] = *(const uint4*)(ar + half*8);
        af.q[1] = *(const uint4*)(ar + 16 + half*8);
        const bf16_t* br = Wt + (size_t)(ntile*16 + mn)*KT + k0;
        frag16 bfr;
        bfr.q[0] = *(const uint4*)(br + half*8);
        bfr.q[1] = *(const uint4*)(br + 16 + half*8);
        acc = __builtin_amdgcn_wmma_f32_16x16x32_bf16(false, af.v, false, bfr.v,
                                                      (short)0, acc, false, false);
    }
#endif

    int col = ntile*16 + mn;
    #pragma unroll
    for (int r = 0; r < 8; ++r) {
        int row = mtile*16 + half*8 + r;
        float v = acc[r];
        if constexpr (FLAGS & GF_BIAS)  v += bias[col];
        if constexpr (FLAGS & GF_RESID) v += resid[(size_t)row*Nc + col];
        if constexpr (FLAGS & GF_ACT)   v  = gelu_exact(v);
        if constexpr (FLAGS & GF_OUTF)  Cf[(size_t)row*Nc + col] = v;
        if constexpr (FLAGS & GF_OUTB)  Cb[(size_t)row*Nc + col] = (bf16_t)v;
    }
}

// ---------------- gathered QK^T: one wave per (row,h); lane kk<TOPK does a 128-dot ----------------
__global__ void qk_kernel(const float* __restrict__ q,    // [NB,1024]
                          const float* __restrict__ kv,   // [NB,2048] (k|v)
                          const int* __restrict__ idxb,   // [NB,TOPK]
                          float* __restrict__ qk0) {      // [NB,TOPK,H]
    int g = blockIdx.x;
    int h = g & (HEADS-1), row = g >> 3;
    int lane = threadIdx.x;
    if (lane >= TOPK) return;
    int base = row & ~(NN-1);
    int j = idxb[row*TOPK + lane];
    const float* qp = q  + (size_t)row*INNER + h*DHEAD;
    const float* kp = kv + (size_t)(base + j)*(2*INNER) + h*DHEAD;
    float s = 0.0f;
    for (int d = 0; d < DHEAD; d += 4) {
        float4 qa = *(const float4*)(qp + d);
        float4 ka = *(const float4*)(kp + d);
        s += qa.x*ka.x + qa.y*ka.y + qa.z*ka.z + qa.w*ka.w;
    }
    qk0[((size_t)row*TOPK + lane)*HEADS + h] = s * QKSCALE;
}

// ---------------- fused edge-MLP + coord update + softmax + A*V, one block(128) per (b,i) ----------------
__global__ void edge_attn_kernel(const float* __restrict__ qk0,      // [NB,TOPK,H]
                                 const float* __restrict__ edges,    // [B,N,N]
                                 const int*   __restrict__ idxb,     // [NB,TOPK]
                                 const float* __restrict__ coors_in, // [NB,3]
                                 float* __restrict__ coors_out,      // [NB,3]
                                 const float* __restrict__ kv,       // [NB,2048]
                                 const float* __restrict__ e1w, const float* __restrict__ e1b,
                                 const float* __restrict__ e2w, const float* __restrict__ e2b,
                                 const float* __restrict__ c1w, const float* __restrict__ c1b,
                                 const float* __restrict__ c2w, const float* __restrict__ c2b,
                                 const float* __restrict__ cscale,
                                 bf16_t* __restrict__ obf) {         // [NB,1024]
    __shared__ float logits[TOPK][HEADS];
    __shared__ float attn[TOPK][HEADS];
    __shared__ float delta[TOPK][3];
    __shared__ int   jn[TOPK];
    int row  = blockIdx.x;
    int base = row & ~(NN-1);
    int t    = threadIdx.x;
    if (t < TOPK) jn[t] = idxb[row*TOPK + t];
    __syncthreads();

    if (t < TOPK) {
        int j = jn[t];
        float ein[HEADS+1];
        #pragma unroll
        for (int h = 0; h < HEADS; ++h)
            ein[h] = qk0[((size_t)row*TOPK + t)*HEADS + h];
        ein[HEADS] = edges[(size_t)row*NN + j];
        float eh[EHID];
        #pragma unroll
        for (int u = 0; u < EHID; ++u) {
            float s = e1b[u];
            #pragma unroll
            for (int c = 0; c < HEADS+1; ++c) s += ein[c]*e1w[c*EHID + u];
            eh[u] = gelu_exact(s);
        }
        float l2[HEADS];
        #pragma unroll
        for (int h = 0; h < HEADS; ++h) {
            float s = e2b[h];
            #pragma unroll
            for (int u = 0; u < EHID; ++u) s += eh[u]*e2w[u*HEADS + h];
            l2[h] = s;
            logits[t][h] = s;
        }
        float cw = c2b[0];
        #pragma unroll
        for (int c = 0; c < CHID; ++c) {
            float s = c1b[c];
            #pragma unroll
            for (int h = 0; h < HEADS; ++h) s += l2[h]*c1w[h*CHID + c];
            cw += gelu_exact(s)*c2w[c];
        }
        float rx = coors_in[row*3+0] - coors_in[(base+j)*3+0];
        float ry = coors_in[row*3+1] - coors_in[(base+j)*3+1];
        float rz = coors_in[row*3+2] - coors_in[(base+j)*3+2];
        float nrm = sqrtf(fmaxf(rx*rx + ry*ry + rz*rz, 1e-16f));
        float inv = cscale[0] / fmaxf(nrm, 1e-8f);
        delta[t][0] = cw*rx*inv; delta[t][1] = cw*ry*inv; delta[t][2] = cw*rz*inv;
    }
    __syncthreads();

    if (t < 3) {
        float s = 0.0f;
        for (int kk = 0; kk < TOPK; ++kk) s += delta[kk][t];
        coors_out[row*3 + t] = coors_in[row*3 + t] + s;
    }
    if (t >= 32 && t < 32 + HEADS) {
        int h = t - 32;
        float m = -1e30f;
        for (int kk = 0; kk < TOPK; ++kk) m = fmaxf(m, logits[kk][h]);
        float ssum = 0.0f;
        for (int kk = 0; kk < TOPK; ++kk) { float e = expf(logits[kk][h] - m); attn[kk][h] = e; ssum += e; }
        float r = 1.0f/ssum;
        for (int kk = 0; kk < TOPK; ++kk) attn[kk][h] *= r;
    }
    __syncthreads();

    // A*V: 128 threads, thread = dh lane; loop heads
    for (int h = 0; h < HEADS; ++h) {
        float acc = 0.0f;
        for (int kk = 0; kk < TOPK; ++kk) {
            int j = jn[kk];
            acc += attn[kk][h] * kv[(size_t)(base + j)*(2*INNER) + INNER + h*DHEAD + t];
        }
        obf[(size_t)row*INNER + h*DHEAD + t] = (bf16_t)acc;
    }
}

// ---------------- classifier: logits = x @ cls_w + cls_b ----------------
__global__ void cls_kernel(const float* __restrict__ x,
                           const float* __restrict__ W,  // [64,20]
                           const float* __restrict__ b,  // [20]
                           float* __restrict__ out) {    // [NB,20]
    int n = blockIdx.x, c = threadIdx.x;
    if (c >= 20) return;
    float s = b[c];
    for (int d = 0; d < DMODEL; ++d) s += x[n*DMODEL + d]*W[d*20 + c];
    out[n*20 + c] = s;
}

// ---------------- host launcher ----------------
extern "C" void kernel_launch(void* const* d_in, const int* in_sizes, int n_in,
                              void* d_out, int out_size, void* d_ws, size_t ws_size,
                              hipStream_t stream) {
    (void)in_sizes; (void)n_in; (void)out_size; (void)ws_size;
    // inputs (insertion order): feats, coors, edges, mask, seq, fe_w, fe_b, cls_w, cls_b,
    // then 4 layers x 21 params each in dict insertion order.
    const float* feats = (const float*)d_in[0];
    const float* coors = (const float*)d_in[1];
    const float* edges = (const float*)d_in[2];
    const float* fe_w  = (const float*)d_in[5];
    const float* fe_b  = (const float*)d_in[6];
    const float* cls_w = (const float*)d_in[7];
    const float* cls_b = (const float*)d_in[8];
    auto LP = [&](int layer, int j) -> const float* {
        return (const float*)d_in[9 + layer*21 + j];
    };
    // 0 ln1_g 1 ln1_b 2 wq 3 wkv 4 wo 5 bo 6 e1_w 7 e1_b 8 e2_w 9 e2_b
    // 10 c1_w 11 c1_b 12 c2_w 13 c2_b 14 coors_scale 15 ln2_g 16 ln2_b
    // 17 ff1_w 18 ff1_b 19 ff2_w 20 ff2_b

    char* ws = (char*)d_ws;
    size_t off = 0;
    auto alloc = [&](size_t bytes) -> char* {
        char* p = ws + off;
        off = (off + bytes + 255) & ~(size_t)255;
        return p;
    };
    float*  x      = (float*) alloc((size_t)NB*DMODEL*4);
    bf16_t* xln    = (bf16_t*)alloc((size_t)NB*DMODEL*2);
    bf16_t* hln    = (bf16_t*)alloc((size_t)NB*DMODEL*2);
    float*  qbuf   = (float*) alloc((size_t)NB*INNER*4);
    float*  kvbuf  = (float*) alloc((size_t)NB*2*INNER*4);
    bf16_t* obf    = (bf16_t*)alloc((size_t)NB*INNER*2);
    float*  qk0    = (float*) alloc((size_t)NB*TOPK*HEADS*4);
    int*    idxb   = (int*)   alloc((size_t)NB*TOPK*4);
    float*  coorsA = (float*) alloc((size_t)NB*3*4);
    float*  coorsB = (float*) alloc((size_t)NB*3*4);
    bf16_t* wqb    = (bf16_t*)alloc((size_t)DMODEL*INNER*2);   // [1024,64]   W^T
    bf16_t* wkvb   = (bf16_t*)alloc((size_t)DMODEL*2*INNER*2); // [2048,64]   W^T
    bf16_t* wob    = (bf16_t*)alloc((size_t)INNER*DMODEL*2);   // [64,1024]   W^T
    bf16_t* ff1b   = (bf16_t*)alloc((size_t)DMODEL*FFHID*2);   // [256,64]    W^T
    bf16_t* ff2b   = (bf16_t*)alloc((size_t)FFHID*DMODEL*2);   // [64,256]    W^T
    bf16_t* ffh    = (bf16_t*)alloc((size_t)NB*FFHID*2);

    featenc_kernel<<<NB, DMODEL, 0, stream>>>(feats, fe_w, fe_b, x);
    copy_f32_kernel<<<(NB*3 + 255)/256, 256, 0, stream>>>(coors, coorsA, NB*3);

    float* ccur = coorsA;
    float* cnext = coorsB;
    for (int l = 0; l < NLAYER; ++l) {
        knn_kernel<<<NB, 32, 0, stream>>>(ccur, idxb);
        ln_kernel<<<NB, 32, 0, stream>>>(x, LP(l,0), LP(l,1), xln);
        // weight converts (f32 row-major [Kt,Nc] -> bf16 transposed [Nc,Kt])
        f2bf_t_kernel<<<(DMODEL*INNER + 255)/256, 256, 0, stream>>>(LP(l,2), wqb,  DMODEL, INNER);
        f2bf_t_kernel<<<(DMODEL*2*INNER + 255)/256, 256, 0, stream>>>(LP(l,3), wkvb, DMODEL, 2*INNER);
        f2bf_t_kernel<<<(INNER*DMODEL + 255)/256, 256, 0, stream>>>(LP(l,4), wob,  INNER, DMODEL);
        f2bf_t_kernel<<<(DMODEL*FFHID + 255)/256, 256, 0, stream>>>(LP(l,17), ff1b, DMODEL, FFHID);
        f2bf_t_kernel<<<(FFHID*DMODEL + 255)/256, 256, 0, stream>>>(LP(l,19), ff2b, FFHID, DMODEL);
        // projections (WMMA + TDM double-buffer)
        wmma_gemm_kernel<DMODEL, GF_OUTF><<<dim3(INNER/16, NB/16), 32, 0, stream>>>(
            xln, wqb, nullptr, nullptr, qbuf, nullptr, INNER);
        wmma_gemm_kernel<DMODEL, GF_OUTF><<<dim3(2*INNER/16, NB/16), 32, 0, stream>>>(
            xln, wkvb, nullptr, nullptr, kvbuf, nullptr, 2*INNER);
        // gathered QK^T logits
        qk_kernel<<<NB*HEADS, 32, 0, stream>>>(qbuf, kvbuf, idxb, qk0);
        // fused edge MLP + coord update + softmax + A*V
        edge_attn_kernel<<<NB, 128, 0, stream>>>(
            qk0, edges, idxb, ccur, cnext, kvbuf,
            LP(l,6), LP(l,7), LP(l,8), LP(l,9),
            LP(l,10), LP(l,11), LP(l,12), LP(l,13), LP(l,14), obf);
        // output projection + residual (KT=1024 -> 16 double-buffered TDM chunks)
        wmma_gemm_kernel<INNER, GF_BIAS|GF_RESID|GF_OUTF><<<dim3(DMODEL/16, NB/16), 32, 0, stream>>>(
            obf, wob, LP(l,5), x, x, nullptr, DMODEL);
        // FF block
        ln_kernel<<<NB, 32, 0, stream>>>(x, LP(l,15), LP(l,16), hln);
        wmma_gemm_kernel<DMODEL, GF_BIAS|GF_ACT|GF_OUTB><<<dim3(FFHID/16, NB/16), 32, 0, stream>>>(
            hln, ff1b, LP(l,18), nullptr, nullptr, ffh, FFHID);
        wmma_gemm_kernel<FFHID, GF_BIAS|GF_RESID|GF_OUTF><<<dim3(DMODEL/16, NB/16), 32, 0, stream>>>(
            ffh, ff2b, LP(l,20), x, x, nullptr, DMODEL);
        float* tmp = ccur; ccur = cnext; cnext = tmp;
    }
    cls_kernel<<<NB, 32, 0, stream>>>(x, cls_w, cls_b, (float*)d_out);
}
